// HENAWModel_72387378806840
// MI455X (gfx1250) — compile-verified, matrix operand order checked by reading, not run on hardware
//
#include <hip/hip_runtime.h>
#include <math.h>

// ---------------------------------------------------------------------------
// CDNA5 (gfx1250) WMMA / TDM types
// ---------------------------------------------------------------------------
typedef _Float16 v16h __attribute__((ext_vector_type(16)));
typedef _Float16 h8v  __attribute__((ext_vector_type(8)));
typedef float    v8f  __attribute__((ext_vector_type(8)));
typedef unsigned int u32x4 __attribute__((ext_vector_type(4)));
typedef int      i32x8 __attribute__((ext_vector_type(8)));
typedef int      i32x4 __attribute__((ext_vector_type(4)));

#define BN_SCALE 0.9999950000374997f   // 1/sqrt(1+1e-5)

// ----- LDS layout (units: _Float16 elements) -------------------------------
// Weight panels (fragment-linear f16):
#define OFF_W1   0                      // 16 frags : ft_w1  K 9->32 pad, N=256
#define OFF_W2   8192                   // 64 frags : ft_w2  K=256, N=128
#define OFF_WH   40960                  // 96 frags : heads  K 176->192, N=256
#define OFF_SW   90112                  // 30 small frags (enc/att/semb)
#define W_HALFS  105472                 // total weight halfs (210944 B)
// Per-wave scratch (2 waves):
#define WAVE_SZ  19968
#define OFF_WAVE W_HALFS
#define A_EMB   0
#define A_QKV   2560
#define A_OBUF  6656
#define A_ATT   8192
#define A_SYS   9728
#define A_H1    0
#define A_HH    4096

#define SMEM_HALFS (W_HALFS + 2 * WAVE_SZ)     // 145408
#define SMEM_BYTES (SMEM_HALFS * 2)            // 290816 B < 320KB WGP LDS

static __device__ inline v8f zero_v8f() {
    v8f z;
#pragma unroll
    for (int i = 0; i < 8; ++i) z[i] = 0.0f;
    return z;
}

static __device__ inline v8f wmma_f16(v16h a, v16h b, v8f c) {
    return __builtin_amdgcn_wmma_f32_16x16x32_f16(false, a, false, b, (short)0, c, false, false);
}

// A-fragment (16x32 window of a row-major f16 LDS tile, row stride ldk halfs).
static __device__ inline v16h lds_load_a(const _Float16* base, int ldk, int lane) {
    const int m = lane & 15;
    const int o = (lane >> 4) << 3;
    const _Float16* p = base + m * ldk + o;
    h8v lo = *(const h8v*)(p);
    h8v hi = *(const h8v*)(p + 16);
    v16h r;
#pragma unroll
    for (int i = 0; i < 8; ++i) { r[i] = lo[i]; r[i + 8] = hi[i]; }
    return r;
}

// B-fragment stored fragment-linear: 32 lanes x 16 halfs contiguous.
static __device__ inline v16h lds_load_b(const _Float16* frag, int lane) {
    return ((const v16h*)frag)[lane];
}

// ---------------------------------------------------------------------------
// Kernel parameters
// ---------------------------------------------------------------------------
struct HPar {
    const float *x, *age;
    const float *run_mean, *run_var, *aw_w1, *aw_b1, *aw_w2, *aw_b2;
    const float *enc_w1, *enc_b1, *enc_w2, *enc_b2;
    const float *g_wq[3], *g_bq[3], *g_wk[3], *g_bk[3];
    const float *g_wv[3], *g_bv[3], *g_wo[3], *g_bo[3];
    const float *g_sw[3], *g_sb[3];
    const float *x_wq, *x_bq, *x_wk, *x_bk, *x_wv, *x_bv, *x_wo, *x_bo;
    const float *ft_w1, *ft_b1, *bn1_g, *bn1_b;
    const float *ft_w2, *ft_b2, *bn2_g, *bn2_b;
    const float *age_w1, *age_b1, *age_w2, *age_b2;
    const float *mort_w1, *mort_b1, *mort_w2, *mort_b2;
    const float *morb_w1, *morb_b1, *morb_w2, *morb_b2;
    const _Float16* pws;        // pre-packed weights in d_ws (fragment-linear)
    float* out;
    int B;
    int use_ws;
};

// ---------------------------------------------------------------------------
// Fragment-linear weight packing (element e -> f16 value)
// K(lane,v,h) = 16*(v>>2) + 8*(lane>>4) + 2*(v&3) + h ; N(lane) = lane&15
// ---------------------------------------------------------------------------
static __device__ inline void fragdec(int e, int KT, int& k, int& n) {
    int frag = e >> 9, w = e & 511;
    int nt = frag / KT, kt = frag - nt * KT;
    int lane = w >> 4, idx = w & 15;
    int v = idx >> 1, hb = idx & 1;
    int kl = ((v >> 2) << 4) | (((lane >> 4) & 1) << 3) | ((v & 3) << 1) | hb;
    k = kt * 32 + kl;
    n = nt * 16 + (lane & 15);
}

static __device__ _Float16 pack_element(const HPar& P, int e) {
    float val = 0.0f;
    if (e < OFF_W2) {                       // ft_w1: K=9 real, N=256, KT=1
        int k, n; fragdec(e - OFF_W1, 1, k, n);
        if (k < 9) val = P.ft_w1[k * 256 + n];
    } else if (e < OFF_WH) {                // ft_w2: K=256, N=128, KT=8
        int k, n; fragdec(e - OFF_W2, 8, k, n);
        val = P.ft_w2[k * 128 + n];
    } else if (e < OFF_SW) {                // heads: K=176 real, N=256, KT=6
        int k, n; fragdec(e - OFF_WH, 6, k, n);
        if (k < 176) {
            if (n < 64)       val = P.age_w1[k * 64 + n];
            else if (n < 128) val = P.mort_w1[k * 64 + (n - 64)];
            else {
                int hd = (n - 128) >> 5, c = (n - 128) & 31;
                val = P.morb_w1[(hd * 176 + k) * 32 + c];
            }
        }
    } else {                                // 30 small 16x16-ish frags
        int r = e - OFF_SW;
        int fid = r >> 9;
        int k, n; fragdec(r & 511, 1, k, n);
        const float* src; int kreal, kbase = 0;
        if (fid < 9)       { src = P.enc_w2 + fid * 256; kreal = 16; }
        else if (fid < 21) { int g = (fid - 9) >> 2, j = (fid - 9) & 3;
                             src = (j == 0) ? P.g_wq[g] : (j == 1) ? P.g_wk[g]
                                 : (j == 2) ? P.g_wv[g] : P.g_wo[g];
                             kreal = 16; }
        else if (fid < 25) { int j = fid - 21;
                             src = (j == 0) ? P.x_wq : (j == 1) ? P.x_wk
                                 : (j == 2) ? P.x_wv : P.x_wo;
                             kreal = 16; }
        else if (fid < 27) { src = P.g_sw[fid - 25]; kreal = 32; }
        else               { src = P.g_sw[2]; kreal = 80; kbase = (fid - 27) * 32; }
        int kk = kbase + k;
        if (kk < kreal) val = src[kk * 16 + n];
    }
    return (_Float16)val;
}

__global__ void henaw_pack(HPar P, _Float16* dst) {
    int e = blockIdx.x * 256 + threadIdx.x;
    if (e < W_HALFS) dst[e] = pack_element(P, e);
}

// ---------------------------------------------------------------------------
// One 16x16 projection: D = A(16x32 window)*B + bias -> f16 store
// (B fragment has zero K-rows 16..31, so A over-read of finite data is safe)
// ---------------------------------------------------------------------------
static __device__ inline void project(const _Float16* in, int ldin, int c0,
                                      const _Float16* bfrag, const float* bias,
                                      _Float16* out, int ldout, int c1, int lane) {
    v16h a = lds_load_a(in + c0, ldin, lane);
    v16h b = lds_load_b(bfrag, lane);
    v8f d = wmma_f16(a, b, zero_v8f());
    const int nl = lane & 15, mb = (lane >> 4) << 3;
    const float bs = bias[nl];
#pragma unroll
    for (int r = 0; r < 8; ++r)
        out[(mb + r) * ldout + c1 + nl] = (_Float16)(d[r] + bs);
}

// ---------------------------------------------------------------------------
// VALU softmax attention core (D=16, NH=4, dh=4), 16 rows x 2 head-pairs.
// qkv layout: [row][t*48 + {q:0, k:16, v:32} + h*4+d]
// ---------------------------------------------------------------------------
template <int T>
static __device__ void attn_valu(const _Float16* qkv, _Float16* ob, int lane) {
    const int W = T * 48, WO = T * 16 + 16;
    const int row = lane & 15;
    const int h0 = (lane >> 4) * 2;
    const _Float16* rq = qkv + row * W;
#pragma unroll
    for (int hh = 0; hh < 2; ++hh) {
        const int h = h0 + hh;
#pragma unroll
        for (int t = 0; t < T; ++t) {
            float sc[T], mx = -1e30f;
#pragma unroll
            for (int s = 0; s < T; ++s) {
                float a = 0.0f;
#pragma unroll
                for (int d = 0; d < 4; ++d)
                    a += (float)rq[t * 48 + h * 4 + d] * (float)rq[s * 48 + 16 + h * 4 + d];
                a *= 0.5f;
                sc[s] = a; mx = fmaxf(mx, a);
            }
            float sum = 0.0f;
#pragma unroll
            for (int s = 0; s < T; ++s) { sc[s] = expf(sc[s] - mx); sum += sc[s]; }
            float inv = 1.0f / sum;
#pragma unroll
            for (int d = 0; d < 4; ++d) {
                float acc = 0.0f;
#pragma unroll
                for (int s = 0; s < T; ++s) acc += sc[s] * (float)rq[s * 48 + 32 + h * 4 + d];
                ob[row * WO + t * 16 + h * 4 + d] = (_Float16)(acc * inv);
            }
        }
    }
}

// ---------------------------------------------------------------------------
// One biomarker group: qkv -> attention -> output proj -> system embedding
// ---------------------------------------------------------------------------
template <int T>
static __device__ void do_group(const HPar& P, int g, const int (&idx)[T],
                                const _Float16* FRS, int fb, int sf, int sembKT,
                                const _Float16* embbuf, _Float16* qkvbuf,
                                _Float16* obuf, _Float16* attbuf,
                                _Float16* sysbuf, int lane) {
    const int WQ = T * 48, WO = T * 16 + 16;
    const int nl = lane & 15, mb = (lane >> 4) << 3;
#pragma unroll
    for (int t = 0; t < T; ++t) {
        project(embbuf, 160, idx[t] * 16, FRS + (fb + 0) * 512, P.g_bq[g], qkvbuf, WQ, t * 48 +  0, lane);
        project(embbuf, 160, idx[t] * 16, FRS + (fb + 1) * 512, P.g_bk[g], qkvbuf, WQ, t * 48 + 16, lane);
        project(embbuf, 160, idx[t] * 16, FRS + (fb + 2) * 512, P.g_bv[g], qkvbuf, WQ, t * 48 + 32, lane);
    }
    __syncthreads();
    attn_valu<T>(qkvbuf, obuf, lane);
#pragma unroll
    for (int r = 0; r < 8; ++r) obuf[(mb + r) * WO + T * 16 + nl] = (_Float16)0.0f;
    __syncthreads();
#pragma unroll
    for (int t = 0; t < T; ++t)
        project(obuf, WO, t * 16, FRS + (fb + 3) * 512, P.g_bo[g], attbuf, 96, t * 16, lane);
#pragma unroll
    for (int r = 0; r < 8; ++r) attbuf[(mb + r) * 96 + 80 + nl] = (_Float16)0.0f;
    __syncthreads();
    v8f acc = zero_v8f();
#pragma unroll
    for (int kt = 0; kt < sembKT; ++kt) {
        v16h a = lds_load_a(attbuf + kt * 32, 96, lane);
        v16h b = lds_load_b(FRS + (sf + kt) * 512, lane);
        acc = wmma_f16(a, b, acc);
    }
    const float bs = P.g_sb[g][nl];
#pragma unroll
    for (int r = 0; r < 8; ++r)
        sysbuf[(mb + r) * 64 + g * 16 + nl] = (_Float16)(acc[r] + bs);
}

// ---------------------------------------------------------------------------
__global__ __launch_bounds__(64, 1) void henaw_main(HPar P) {
    extern __shared__ _Float16 smem[];
    const int tid  = threadIdx.x;
    const int lane = tid & 31;
    const int wave = tid >> 5;
    const int nl = lane & 15, mb = (lane >> 4) << 3;

    // ---- stage packed weights into LDS ------------------------------------
    if (P.use_ws) {
#if __has_builtin(__builtin_amdgcn_tensor_load_to_lds) && __has_builtin(__builtin_amdgcn_s_wait_tensorcnt)
        // Tensor Data Mover: DMA the whole 210944-B packed blob to LDS offset 0.
        // D# group0: count=1 | lds_addr=0 | global_addr | type=2 ("image")
        // D# group1: data_size=3 (8B) | tensor_dim0=tile_dim0=26368 | dims1=1
        if (tid == 0) {
            unsigned long long ga = (unsigned long long)P.pws;
            const unsigned ND = (unsigned)(W_HALFS * 2 / 8);   // 26368 8-byte elems
            u32x4 g0;
            g0[0] = 1u;                                        // count=1
            g0[1] = 0u;                                        // lds_addr = 0
            g0[2] = (unsigned)(ga & 0xffffffffu);              // global_addr[31:0]
            g0[3] = (unsigned)((ga >> 32) & 0x01ffffffu) | 0x80000000u; // [56:32]|type=2
            i32x8 g1;
            g1[0] = 0x30000;                                   // data_size=3 (8B)
            g1[1] = (int)((ND & 0xffffu) << 16);               // tensor_dim0[15:0]
            g1[2] = (int)((ND >> 16) | (1u << 16));            // tensor_dim0[31:16]|tensor_dim1=1
            g1[3] = (int)(ND << 16);                           // tile_dim0
            g1[4] = 1;                                         // tile_dim1=1, tile_dim2=0
            g1[5] = (int)ND;                                   // tensor_dim0_stride lo
            g1[6] = 0;
            g1[7] = 0;
            i32x4 z4 = {0, 0, 0, 0};
            i32x8 z8 = {0, 0, 0, 0, 0, 0, 0, 0};
            __builtin_amdgcn_tensor_load_to_lds(g0, g1, z4, z4, z8, 0);
        }
        __builtin_amdgcn_s_wait_tensorcnt(0);
#else
        const uint4* s4 = (const uint4*)P.pws;
        uint4* d4 = (uint4*)smem;
#pragma unroll 4
        for (int i = tid; i < (W_HALFS / 8); i += 64) d4[i] = s4[i];
#endif
    } else {
#pragma unroll 1
        for (int e = tid; e < W_HALFS; e += 64) smem[e] = pack_element(P, e);
    }
    __syncthreads();

    _Float16* FRS  = smem + OFF_SW;
    _Float16* wb   = smem + OFF_WAVE + wave * WAVE_SZ;
    _Float16* xwA  = wb;
    _Float16* encA = wb + 1024;
    _Float16* cbufT = wb + 6144;
    _Float16* AR   = wb + 9216;
    _Float16* embbuf = AR + A_EMB;
    _Float16* qkvbuf = AR + A_QKV;
    _Float16* obuf   = AR + A_OBUF;
    _Float16* attbuf = AR + A_ATT;
    _Float16* sysbuf = AR + A_SYS;
    _Float16* h1b    = AR + A_H1;
    _Float16* hhb    = AR + A_HH;

    const int nchunks = (P.B + 63) >> 6;
#pragma unroll 1
    for (int c = blockIdx.x; c < nchunks; c += gridDim.x) {
        // ---- phase A: per-row gating + encoder hidden (VALU, 1 row/lane) --
        {
            int row = c * 64 + tid;
            if (row >= P.B) row = P.B - 1;
            __builtin_prefetch(P.x + (size_t)row * 9, 0, 3);

            float x9[9];
#pragma unroll
            for (int i = 0; i < 9; ++i) x9[i] = P.x[(size_t)row * 9 + i];
            const float agev = P.age[row];

            float bw[9]; float mx = -1e30f;
#pragma unroll
            for (int i = 0; i < 9; ++i) {
                float cv = sqrtf(P.run_var[i]) / (fabsf(P.run_mean[i]) + 1e-8f);
                bw[i] = -2.0f * cv; mx = fmaxf(mx, bw[i]);
            }
            float s = 0.0f;
#pragma unroll
            for (int i = 0; i < 9; ++i) { bw[i] = expf(bw[i] - mx); s += bw[i]; }
            float inv = 1.0f / s;

            const float an = (agev - 50.0f) / 20.0f;
            float h32[32];
#pragma unroll
            for (int j = 0; j < 32; ++j) h32[j] = fmaxf(an * P.aw_w1[j] + P.aw_b1[j], 0.0f);
            float lg[9]; mx = -1e30f;
#pragma unroll
            for (int i = 0; i < 9; ++i) {
                float a = P.aw_b2[i];
#pragma unroll
                for (int j = 0; j < 32; ++j) a += h32[j] * P.aw_w2[j * 9 + i];
                lg[i] = a; mx = fmaxf(mx, a);
            }
            float s2 = 0.0f;
#pragma unroll
            for (int i = 0; i < 9; ++i) { lg[i] = expf(lg[i] - mx); s2 += lg[i]; }
            float inv2 = 1.0f / s2;
            float wv9[9]; float s3 = 0.0f;
#pragma unroll
            for (int i = 0; i < 9; ++i) { wv9[i] = bw[i] * inv * lg[i] * inv2; s3 += wv9[i]; }
            float inv3 = 1.0f / (s3 + 1e-8f);

#pragma unroll
            for (int k = 0; k < 32; ++k)
                xwA[lane * 32 + k] = (k < 9) ? (_Float16)(x9[k] * wv9[k] * inv3) : (_Float16)0.0f;

#pragma unroll
            for (int n = 0; n < 9; ++n)
#pragma unroll
                for (int d = 0; d < 16; ++d) {
                    float hv = fmaxf(x9[n] * P.enc_w1[n * 16 + d] + P.enc_b1[n * 16 + d], 0.0f);
                    encA[lane * 160 + n * 16 + d] = (_Float16)hv;
                }
#pragma unroll
            for (int j = 0; j < 16; ++j) encA[lane * 160 + 144 + j] = (_Float16)0.0f;
        }
        __syncthreads();

        // ---- two 16-row M-tiles per wave ----------------------------------
#pragma unroll 1
        for (int t = 0; t < 2; ++t) {
#pragma unroll
            for (int r = 0; r < 8; ++r) {
                embbuf[(mb + r) * 160 + 144 + nl] = (_Float16)0.0f;
                sysbuf[(mb + r) * 64 + 48 + nl]   = (_Float16)0.0f;
                cbufT[(mb + r) * 192 + 176 + nl]  = (_Float16)0.0f;
            }
            const _Float16* encT = encA + t * 16 * 160;
#pragma unroll
            for (int n = 0; n < 9; ++n)
                project(encT, 160, n * 16, FRS + n * 512, P.enc_b2 + n * 16,
                        embbuf, 160, n * 16, lane);
            __syncthreads();

            { const int i0[2] = {0, 4};
              do_group<2>(P, 0, i0, FRS,  9, 25, 1, embbuf, qkvbuf, obuf, attbuf, sysbuf, lane); }
            { const int i1[2] = {1, 6};
              do_group<2>(P, 1, i1, FRS, 13, 26, 1, embbuf, qkvbuf, obuf, attbuf, sysbuf, lane); }
            { const int i2[5] = {2, 3, 5, 7, 8};
              do_group<5>(P, 2, i2, FRS, 17, 27, 3, embbuf, qkvbuf, obuf, attbuf, sysbuf, lane); }
            __syncthreads();

#pragma unroll
            for (int tt = 0; tt < 3; ++tt) {
                project(sysbuf, 64, tt * 16, FRS + 21 * 512, P.x_bq, qkvbuf, 144, tt * 48 +  0, lane);
                project(sysbuf, 64, tt * 16, FRS + 22 * 512, P.x_bk, qkvbuf, 144, tt * 48 + 16, lane);
                project(sysbuf, 64, tt * 16, FRS + 23 * 512, P.x_bv, qkvbuf, 144, tt * 48 + 32, lane);
            }
            __syncthreads();
            attn_valu<3>(qkvbuf, obuf, lane);
#pragma unroll
            for (int r = 0; r < 8; ++r) obuf[(mb + r) * 64 + 48 + nl] = (_Float16)0.0f;
            __syncthreads();
#pragma unroll
            for (int tt = 0; tt < 3; ++tt)
                project(obuf, 64, tt * 16, FRS + 24 * 512, P.x_bo, cbufT, 192, tt * 16, lane);
            __syncthreads();

            // ---- backend: h1 -> h2 -> fused heads (WMMA) ------------------
            {
                v16h a = lds_load_a(xwA + t * 16 * 32, 32, lane);
#pragma unroll
                for (int nt = 0; nt < 16; ++nt) {
                    v16h b = lds_load_b(smem + OFF_W1 + nt * 512, lane);
                    v8f d = wmma_f16(a, b, zero_v8f());
                    int n = nt * 16 + nl;
                    float b1 = P.ft_b1[n], g = P.bn1_g[n] * BN_SCALE, bb = P.bn1_b[n];
#pragma unroll
                    for (int r = 0; r < 8; ++r)
                        h1b[(mb + r) * 256 + n] = (_Float16)(fmaxf(d[r] + b1, 0.0f) * g + bb);
                }
            }
            __syncthreads();
#pragma unroll 1
            for (int nt = 0; nt < 8; ++nt) {
                v8f acc = zero_v8f();
#pragma unroll
                for (int kt = 0; kt < 8; ++kt) {
                    v16h a = lds_load_a(h1b + kt * 32, 256, lane);
                    v16h b = lds_load_b(smem + OFF_W2 + (nt * 8 + kt) * 512, lane);
                    acc = wmma_f16(a, b, acc);
                }
                int n = nt * 16 + nl;
                float b2 = P.ft_b2[n], g = P.bn2_g[n] * BN_SCALE, bb = P.bn2_b[n];
#pragma unroll
                for (int r = 0; r < 8; ++r)
                    cbufT[(mb + r) * 192 + 48 + n] = (_Float16)(fmaxf(acc[r] + b2, 0.0f) * g + bb);
            }
            __syncthreads();
#pragma unroll 1
            for (int nt = 0; nt < 16; ++nt) {
                v8f acc = zero_v8f();
#pragma unroll
                for (int kt = 0; kt < 6; ++kt) {
                    v16h a = lds_load_a(cbufT + kt * 32, 192, lane);
                    v16h b = lds_load_b(smem + OFF_WH + (nt * 6 + kt) * 512, lane);
                    acc = wmma_f16(a, b, acc);
                }
                int n = nt * 16 + nl;
                float hb;
                if (n < 64)       hb = P.age_b1[n];
                else if (n < 128) hb = P.mort_b1[n - 64];
                else              hb = P.morb_b1[((n - 128) >> 5) * 32 + ((n - 128) & 31)];
#pragma unroll
                for (int r = 0; r < 8; ++r)
                    hhb[(mb + r) * 256 + n] = (_Float16)fmaxf(acc[r] + hb, 0.0f);
            }
            __syncthreads();

            {
                const int row = nl;
                const _Float16* hr = hhb + row * 256;
                int grow = c * 64 + wave * 32 + t * 16 + row;
                if (grow < P.B) {
                    float* op = P.out + (size_t)grow * 6;
                    if (lane < 16) {
                        float ba = P.age_b2[0], mo = P.mort_b2[0];
#pragma unroll
                        for (int j = 0; j < 64; ++j) {
                            ba += (float)hr[j]      * P.age_w2[j];
                            mo += (float)hr[64 + j] * P.mort_w2[j];
                        }
                        op[0] = ba; op[1] = mo;
                    } else {
#pragma unroll
                        for (int hd = 0; hd < 4; ++hd) {
                            float a = P.morb_b2[hd];
#pragma unroll
                            for (int j = 0; j < 32; ++j)
                                a += (float)hr[128 + hd * 32 + j] * P.morb_w2[hd * 32 + j];
                            op[2 + hd] = 1.0f / (1.0f + expf(-a));
                        }
                    }
                }
            }
            __syncthreads();
        }
    }
}

// ---------------------------------------------------------------------------
extern "C" void kernel_launch(void* const* d_in, const int* in_sizes, int n_in,
                              void* d_out, int out_size, void* d_ws, size_t ws_size,
                              hipStream_t stream) {
    (void)n_in; (void)out_size;
    const float* const* F = (const float* const*)d_in;

    HPar P;
    P.x = F[0]; P.age = F[1];
    P.run_mean = F[2]; P.run_var = F[3];
    P.aw_w1 = F[4]; P.aw_b1 = F[5]; P.aw_w2 = F[6]; P.aw_b2 = F[7];
    P.enc_w1 = F[8]; P.enc_b1 = F[9]; P.enc_w2 = F[10]; P.enc_b2 = F[11];
    for (int g = 0; g < 3; ++g) {
        int base = 12 + g * 10;
        P.g_wq[g] = F[base + 0]; P.g_bq[g] = F[base + 1];
        P.g_wk[g] = F[base + 2]; P.g_bk[g] = F[base + 3];
        P.g_wv[g] = F[base + 4]; P.g_bv[g] = F[base + 5];
        P.g_wo[g] = F[base + 6]; P.g_bo[g] = F[base + 7];
        P.g_sw[g] = F[base + 8]; P.g_sb[g] = F[base + 9];
    }
    P.x_wq = F[42]; P.x_bq = F[43]; P.x_wk = F[44]; P.x_bk = F[45];
    P.x_wv = F[46]; P.x_bv = F[47]; P.x_wo = F[48]; P.x_bo = F[49];
    P.ft_w1 = F[50]; P.ft_b1 = F[51]; P.bn1_g = F[52]; P.bn1_b = F[53];
    P.ft_w2 = F[54]; P.ft_b2 = F[55]; P.bn2_g = F[56]; P.bn2_b = F[57];
    P.age_w1 = F[58]; P.age_b1 = F[59]; P.age_w2 = F[60]; P.age_b2 = F[61];
    P.mort_w1 = F[62]; P.mort_b1 = F[63]; P.mort_w2 = F[64]; P.mort_b2 = F[65];
    P.morb_w1 = F[66]; P.morb_b1 = F[67]; P.morb_w2 = F[68]; P.morb_b2 = F[69];
    P.out = (float*)d_out;
    P.B = in_sizes[1];
    P.pws = (const _Float16*)d_ws;
    P.use_ws = (d_ws != nullptr && ws_size >= (size_t)W_HALFS * 2) ? 1 : 0;

    (void)hipFuncSetAttribute((const void*)henaw_main,
                              hipFuncAttributeMaxDynamicSharedMemorySize, SMEM_BYTES);

    if (P.use_ws)
        henaw_pack<<<(W_HALFS + 255) / 256, 256, 0, stream>>>(P, (_Float16*)d_ws);

    int nchunks = (P.B + 63) >> 6;
    int grid = nchunks < 1024 ? nchunks : 1024;
    henaw_main<<<grid, 64, SMEM_BYTES, stream>>>(P);
}
